// UGCNN_85495618994585
// MI455X (gfx1250) — matrix operand order repaired and checked by reading, not compiled, not run on hardware
//
#include <hip/hip_runtime.h>
#include <hip/hip_bf16.h>
#include <stdint.h>

// ---------------------------------------------------------------------------
// Types for CDNA5 WMMA (gfx1250, wave32)
// ---------------------------------------------------------------------------
typedef __bf16 bf16x16 __attribute__((ext_vector_type(16)));
typedef float  f32x8   __attribute__((ext_vector_type(8)));

__device__ __forceinline__ unsigned int f2bf1(float f) {
    unsigned int u = __float_as_uint(f);
    // round-to-nearest-even f32 -> bf16
    return (u + 0x7FFFu + ((u >> 16) & 1u)) >> 16;
}
__device__ __forceinline__ unsigned int pack2bf(float a, float b) {
    return f2bf1(a) | (f2bf1(b) << 16);
}

// ---------------------------------------------------------------------------
// Utility kernels
// ---------------------------------------------------------------------------
__global__ void fill_f32(float* __restrict__ p, float v, long long n) {
    long long i = (long long)blockIdx.x * blockDim.x + threadIdx.x;
    if (i < n) p[i] = v;
}

__global__ void rsqrt_inplace(float* __restrict__ p, int n) {
    int i = blockIdx.x * blockDim.x + threadIdx.x;
    if (i < n) p[i] = rsqrtf(p[i]);
}

// degree accumulation over real edges (self-loops folded into init value 1.0)
__global__ void deg_acc(const long long* __restrict__ dst, float* __restrict__ deg,
                        long long E) {
    long long i = (long long)blockIdx.x * blockDim.x + threadIdx.x;
    long long stride = (long long)gridDim.x * blockDim.x;
    for (; i < E; i += stride) atomicAdd(&deg[dst[i]], 1.0f);
}

// convert W [k][n] f32 -> Wt [n][k] bf16 (128x128)
__global__ void conv_w(const float* __restrict__ W, unsigned short* __restrict__ Wt) {
    int idx = blockIdx.x * 256 + threadIdx.x;   // 0..16383
    int k = idx >> 7, n = idx & 127;
    Wt[n * 128 + k] = (unsigned short)f2bf1(W[k * 128 + n]);
}

// ---------------------------------------------------------------------------
// WMMA GEMM: C[N x 128] = A[N x 128] @ W[128 x 128]   (A f32, W pre-bf16 [n][k])
// block = 256 threads (8 waves), tile = 128 rows x 128 cols
// ---------------------------------------------------------------------------
#define LDS_STRIDE 136   // 128 + 8 bf16 pad: rows land 4 banks apart -> conflict-free b128

__global__ __launch_bounds__(256) void gemm128_bf16(
    const float* __restrict__ A, const unsigned short* __restrict__ Wt,
    float* __restrict__ C, int N)
{
    __shared__ __align__(16) unsigned short As[128 * LDS_STRIDE];
    __shared__ __align__(16) unsigned short Bs[128 * LDS_STRIDE];

    const int t = threadIdx.x;
    const int rowBase  = blockIdx.x * 128;
    const int rowsHere = min(128, N - rowBase);

    // ---- stage A tile (f32 -> bf16) into LDS ----
    #pragma unroll
    for (int i = 0; i < 16; ++i) {
        int o = (t + 256 * i) << 2;           // element offset, step 4, covers 16384
        int r = o >> 7, c = o & 127;
        float4 v = make_float4(0.f, 0.f, 0.f, 0.f);
        if (r < rowsHere)
            v = *(const float4*)(A + (size_t)(rowBase + r) * 128 + c);
        uint2 pp;
        pp.x = pack2bf(v.x, v.y);
        pp.y = pack2bf(v.z, v.w);
        *(uint2*)(&As[r * LDS_STRIDE + c]) = pp;
    }
    // ---- stage Wt (already bf16, [n][k]) into LDS ----
    #pragma unroll
    for (int i = 0; i < 16; ++i) {
        int o = (t + 256 * i) << 2;           // ushort offset, step 4
        int n = o >> 7, k = o & 127;
        uint2 w = *(const uint2*)(Wt + o);
        *(uint2*)(&Bs[n * LDS_STRIDE + k]) = w;
    }
    __syncthreads();

    const int lane  = t & 31;
    const int wv    = t >> 5;
    const int row16 = lane & 15;
    const int half  = lane >> 4;              // 0 for lanes 0-15, 1 for 16-31
    const int wrow  = wv * 16;                // wave's 16-row stripe

    f32x8 acc[8];
    #pragma unroll
    for (int i = 0; i < 8; ++i) acc[i] = (f32x8){0, 0, 0, 0, 0, 0, 0, 0};

    #pragma unroll
    for (int kt = 0; kt < 4; ++kt) {
        const int kb = kt * 32;
        // A fragment: lanes 0-15 -> K kb+0..7 / kb+16..23 ; lanes 16-31 -> +8
        union { uint4 q[2]; bf16x16 v; } ua;
        const unsigned short* ap = &As[(wrow + row16) * LDS_STRIDE + kb + half * 8];
        ua.q[0] = *(const uint4*)(ap);        // K = kb + half*8 .. +7
        ua.q[1] = *(const uint4*)(ap + 16);   // K = kb + 16 + half*8 .. +7

        #pragma unroll
        for (int nt = 0; nt < 8; ++nt) {
            // B fragment: column N = nt*16 + row16, K = kb + half*8 .. +7 / kb + 16 + half*8 .. +7
            union { uint4 q[2]; bf16x16 v; } ub;
            const unsigned short* bp = &Bs[(nt * 16 + row16) * LDS_STRIDE + kb + half * 8];
            ub.q[0] = *(const uint4*)(bp);
            ub.q[1] = *(const uint4*)(bp + 16);
            acc[nt] = __builtin_amdgcn_wmma_f32_16x16x32_bf16(
                false, ua.v, false, ub.v, (short)0, acc[nt], false, false);
        }
    }

    // D layout: VGPR v -> (M = v + half*8, N = row16)
    if (rowBase + wrow < N) {                 // wave-uniform tail guard (N % 16 == 0)
        #pragma unroll
        for (int nt = 0; nt < 8; ++nt) {
            #pragma unroll
            for (int v = 0; v < 8; ++v) {
                int m = rowBase + wrow + half * 8 + v;
                C[(size_t)m * 128 + nt * 16 + row16] = acc[nt][v];
            }
        }
    }
}

// ---------------------------------------------------------------------------
// agg[n] = bias + h[n] * dinv[n]^2   (bias + self-loop term), float4 vectorized
// ---------------------------------------------------------------------------
__global__ void agg_init(const float* __restrict__ h, const float* __restrict__ dinv,
                         const float* __restrict__ bias, float* __restrict__ agg,
                         int nvec /* = N*32 */) {
    int i = blockIdx.x * 256 + threadIdx.x;
    if (i >= nvec) return;
    int n  = i >> 5;
    int fq = i & 31;
    float di = dinv[n];
    float w  = di * di;
    float4 v = ((const float4*)h)[i];
    float4 b = ((const float4*)bias)[fq];
    float4 r;
    r.x = b.x + v.x * w; r.y = b.y + v.y * w;
    r.z = b.z + v.z * w; r.w = b.w + v.w * w;
    ((float4*)agg)[i] = r;
}

// ---------------------------------------------------------------------------
// edge scatter: agg[dst] += h[src] * dinv[src]*dinv[dst]
// one wave per edge: 32 lanes x float4 = full 128-feature row
// ---------------------------------------------------------------------------
__global__ __launch_bounds__(256) void scatter_edges(
    const float* __restrict__ h, const long long* __restrict__ src,
    const long long* __restrict__ dst, const float* __restrict__ dinv,
    float* __restrict__ agg, long long E)
{
    const int lane = threadIdx.x & 31;
    long long w  = (long long)blockIdx.x * 8 + (threadIdx.x >> 5);
    long long nw = (long long)gridDim.x * 8;
    for (long long e = w; e < E; e += nw) {
        long long s = src[e];
        long long d = dst[e];
        float nrm = dinv[s] * dinv[d];
        float4 v = ((const float4*)(h + s * 128))[lane];
        float* ad = agg + d * 128 + lane * 4;
        atomicAdd(ad + 0, v.x * nrm);
        atomicAdd(ad + 1, v.y * nrm);
        atomicAdd(ad + 2, v.z * nrm);
        atomicAdd(ad + 3, v.w * nrm);
    }
}

// ---------------------------------------------------------------------------
// BatchNorm (training-style batch stats) + ReLU
// ---------------------------------------------------------------------------
__global__ __launch_bounds__(128) void bn_stats(const float* __restrict__ x,
                                                float* __restrict__ stats, int N) {
    int f = threadIdx.x;   // 128 features
    float s = 0.f, q = 0.f;
    for (int r = blockIdx.x; r < N; r += gridDim.x) {
        float v = x[(size_t)r * 128 + f];
        s += v; q += v * v;
    }
    atomicAdd(&stats[f], s);
    atomicAdd(&stats[128 + f], q);
}

__global__ __launch_bounds__(128) void bn_finalize(const float* __restrict__ stats,
                                                   const float* __restrict__ gamma,
                                                   const float* __restrict__ beta,
                                                   float* __restrict__ coef, int N) {
    int f = threadIdx.x;
    float inv_n = 1.0f / (float)N;
    float mu  = stats[f] * inv_n;
    float var = fmaxf(stats[128 + f] * inv_n - mu * mu, 0.0f);  // biased variance
    float sc  = gamma[f] * rsqrtf(var + 1e-5f);
    coef[f]       = sc;
    coef[128 + f] = beta[f] - mu * sc;
}

__global__ void bn_apply_relu(float* __restrict__ x, const float* __restrict__ coef,
                              int nvec) {
    int i = blockIdx.x * 256 + threadIdx.x;
    if (i >= nvec) return;
    int fq = i & 31;
    float4 sc = ((const float4*)coef)[fq];
    float4 sh = ((const float4*)(coef + 128))[fq];
    float4 v  = ((float4*)x)[i];
    v.x = fmaxf(v.x * sc.x + sh.x, 0.f);
    v.y = fmaxf(v.y * sc.y + sh.y, 0.f);
    v.z = fmaxf(v.z * sc.z + sh.z, 0.f);
    v.w = fmaxf(v.w * sc.w + sh.w, 0.f);
    ((float4*)x)[i] = v;
}

// ---------------------------------------------------------------------------
// global mean pool (batch sorted; atomics into [G][128] + counts)
// ---------------------------------------------------------------------------
__global__ __launch_bounds__(128) void pool_acc(const float* __restrict__ x,
                                                const long long* __restrict__ batch,
                                                float* __restrict__ psum,
                                                float* __restrict__ cnt, int N) {
    int f = threadIdx.x;
    for (int r = blockIdx.x; r < N; r += gridDim.x) {
        int g = (int)batch[r];
        atomicAdd(&psum[(size_t)g * 128 + f], x[(size_t)r * 128 + f]);
        if (f == 0) atomicAdd(&cnt[g], 1.0f);
    }
}

// out[g][o] = (psum[g]/max(cnt,1)) @ Wout + bout     (128x64, tiny)
__global__ __launch_bounds__(64) void final_gemm(const float* __restrict__ psum,
                                                 const float* __restrict__ cnt,
                                                 const float* __restrict__ Wout,
                                                 const float* __restrict__ bout,
                                                 float* __restrict__ out) {
    __shared__ float p[128];
    int g = blockIdx.x, o = threadIdx.x;
    float c = 1.0f / fmaxf(cnt[g], 1.0f);
    p[o]      = psum[(size_t)g * 128 + o]      * c;
    p[o + 64] = psum[(size_t)g * 128 + o + 64] * c;
    __syncthreads();
    float acc = bout[o];
    #pragma unroll 4
    for (int h = 0; h < 128; ++h) acc = fmaf(p[h], Wout[h * 64 + o], acc);
    out[g * 64 + o] = acc;
}

// ---------------------------------------------------------------------------
// launch
// ---------------------------------------------------------------------------
extern "C" void kernel_launch(void* const* d_in, const int* in_sizes, int n_in,
                              void* d_out, int out_size, void* d_ws, size_t ws_size,
                              hipStream_t stream) {
    const float*     x    = (const float*)d_in[0];
    const long long* ei   = (const long long*)d_in[1];   // [2,E] int64
    const long long* batch= (const long long*)d_in[2];
    const float*     W0   = (const float*)d_in[4];
    const float*     b0   = (const float*)d_in[5];
    const float*     g0   = (const float*)d_in[6];
    const float*     be0  = (const float*)d_in[7];
    const float*     W1   = (const float*)d_in[8];
    const float*     b1   = (const float*)d_in[9];
    const float*     g1   = (const float*)d_in[10];
    const float*     be1  = (const float*)d_in[11];
    const float*     Wout = (const float*)d_in[12];
    const float*     bout = (const float*)d_in[13];
    float*           out  = (float*)d_out;

    const int       N = in_sizes[0] / 128;           // 100000
    const long long E = (long long)in_sizes[1] / 2;  // 1.6M
    const int       G = out_size / 64;               // 128 graphs
    const long long srcOfs = 0, dstOfs = E;
    const long long* src = ei + srcOfs;
    const long long* dst = ei + dstOfs;

    // ---- carve workspace (256B aligned slabs) ----
    char* wp = (char*)d_ws;
    auto carve = [&](size_t bytes) -> void* {
        void* r = (void*)wp;
        wp += (bytes + 255) & ~(size_t)255;
        return r;
    };
    float*          dinv  = (float*)carve((size_t)N * sizeof(float));
    float*          bufH  = (float*)carve((size_t)N * 128 * sizeof(float));
    float*          bufA  = (float*)carve((size_t)N * 128 * sizeof(float));
    unsigned short* Wt0   = (unsigned short*)carve(16384 * sizeof(unsigned short));
    unsigned short* Wt1   = (unsigned short*)carve(16384 * sizeof(unsigned short));
    float*          stats = (float*)carve(256 * sizeof(float));
    float*          coef  = (float*)carve(256 * sizeof(float));
    float*          psum  = (float*)carve((size_t)G * 128 * sizeof(float));
    float*          cnt   = (float*)carve((size_t)G * sizeof(float));
    (void)ws_size; (void)n_in;

    const int nvec = N * 32;                       // float4 count of an NxH buffer
    const int gEl  = (nvec + 255) / 256;           // elementwise grid
    const int gGemm= (N + 127) / 128;

    // ---- prep: bf16-transposed weights, degrees ----
    conv_w<<<64, 256, 0, stream>>>(W0, Wt0);
    conv_w<<<64, 256, 0, stream>>>(W1, Wt1);
    fill_f32<<<(N + 255) / 256, 256, 0, stream>>>(dinv, 1.0f, N);  // self-loop degree
    deg_acc<<<2048, 256, 0, stream>>>(dst, dinv, E);
    rsqrt_inplace<<<(N + 255) / 256, 256, 0, stream>>>(dinv, N);

    // ================= layer 1 =================
    gemm128_bf16<<<gGemm, 256, 0, stream>>>(x, Wt0, bufH, N);            // h = x @ W0
    agg_init<<<gEl, 256, 0, stream>>>(bufH, dinv, b0, bufA, nvec);       // b0 + self-loop
    scatter_edges<<<4096, 256, 0, stream>>>(bufH, src, dst, dinv, bufA, E);
    fill_f32<<<1, 256, 0, stream>>>(stats, 0.0f, 256);
    bn_stats<<<1024, 128, 0, stream>>>(bufA, stats, N);
    bn_finalize<<<1, 128, 0, stream>>>(stats, g0, be0, coef, N);
    bn_apply_relu<<<gEl, 256, 0, stream>>>(bufA, coef, nvec);            // act1 (in place)

    // ================= layer 2 =================
    gemm128_bf16<<<gGemm, 256, 0, stream>>>(bufA, Wt1, bufH, N);         // h = act1 @ W1
    agg_init<<<gEl, 256, 0, stream>>>(bufH, dinv, b1, bufA, nvec);       // bufA reusable now
    scatter_edges<<<4096, 256, 0, stream>>>(bufH, src, dst, dinv, bufA, E);
    fill_f32<<<1, 256, 0, stream>>>(stats, 0.0f, 256);
    bn_stats<<<1024, 128, 0, stream>>>(bufA, stats, N);
    bn_finalize<<<1, 128, 0, stream>>>(stats, g1, be1, coef, N);
    bn_apply_relu<<<gEl, 256, 0, stream>>>(bufA, coef, nvec);

    // ================= pool + head =================
    fill_f32<<<(G * 128 + 255) / 256, 256, 0, stream>>>(psum, 0.0f, (long long)G * 128);
    fill_f32<<<1, 256, 0, stream>>>(cnt, 0.0f, G);
    pool_acc<<<1024, 128, 0, stream>>>(bufA, batch, psum, cnt, N);
    final_gemm<<<G, 64, 0, stream>>>(psum, cnt, Wout, bout, out);
}
